// CausalMultiHeadAttention_51307679318253
// MI455X (gfx1250) — matrix operand order repaired and checked by reading, not compiled, not run on hardware
//
#include <hip/hip_runtime.h>
#include <hip/hip_bf16.h>
#include <math.h>

// ---------------------------------------------------------------------------
// CDNA5 (gfx1250) causal MHA forward, bf16 WMMA with f32 accumulation.
// Wave32 layouts per CDNA5 ISA 7.12.2:
//   A (16x32 bf16): lane m=lane&15, h=lane>>4; v0..3 = K(8h..8h+7), v4..7 = K(16+8h..23+8h)
//   B (32x16 bf16): lane n=lane&15, h; v0..3 = K(8h..8h+7), v4..7 = K(16+8h..23+8h) for column n
//   C/D (16x16 f32): lane n=lane&15, h; reg r holds row M=r+8h, col n
// ---------------------------------------------------------------------------

typedef __bf16 bf16_t;
typedef __bf16 bf8v  __attribute__((ext_vector_type(8)));
typedef __bf16 bf16v __attribute__((ext_vector_type(16)));
typedef float  v8f   __attribute__((ext_vector_type(8)));

#define S_LEN 2048
#define HD 64
#define NH 16
#define BATCH 4
#define DMODEL 1024
#define MROWS 8192          // B*S

__device__ __forceinline__ v8f zero8() {
  v8f z;
#pragma unroll
  for (int i = 0; i < 8; ++i) z[i] = 0.0f;
  return z;
}

__device__ __forceinline__ bf16v ones16() {
  bf16v o;
#pragma unroll
  for (int i = 0; i < 16; ++i) o[i] = (bf16_t)1.0f;
  return o;
}

__device__ __forceinline__ bf8v ld8(const bf16_t* p) {
  return *reinterpret_cast<const bf8v*>(p);   // 16B -> global_load_b128 / ds_load_b128
}

__device__ __forceinline__ bf16v concat8(bf8v lo, bf8v hi) {
  return __builtin_shufflevector(lo, hi, 0,1,2,3,4,5,6,7,8,9,10,11,12,13,14,15);
}

// A operand from row-major [M][K]: rowbase -> element [m][0] for this lane's row m
__device__ __forceinline__ bf16v load_matA(const bf16_t* rowbase, int k0, int h) {
  const bf16_t* p = rowbase + k0 + 8 * h;
  return concat8(ld8(p), ld8(p + 16));
}

// B operand from B^T row-major [N][K]: rowbase -> element [n][0] for this lane's col n
__device__ __forceinline__ bf16v load_matB(const bf16_t* rowbase, int k0, int h) {
  const bf16_t* p = rowbase + k0 + 8 * h;
  return concat8(ld8(p), ld8(p + 16));
}

__device__ __forceinline__ v8f wmma_bf16(bf16v a, bf16v b, v8f c) {
  return __builtin_amdgcn_wmma_f32_16x16x32_bf16(false, a, false, b, (short)0, c,
                                                 false, false);
}

__device__ __forceinline__ float rmax16(float v) {
  v = fmaxf(v, __shfl_xor(v, 1, 32));
  v = fmaxf(v, __shfl_xor(v, 2, 32));
  v = fmaxf(v, __shfl_xor(v, 4, 32));
  v = fmaxf(v, __shfl_xor(v, 8, 32));
  return v;
}

// --------------------------- conversion kernels ----------------------------

__global__ __launch_bounds__(512) void cvt_f32_bf16(const float* __restrict__ src,
                                                    bf16_t* __restrict__ dst, int n) {
  int i = blockIdx.x * 512 + threadIdx.x;
  if (i < n) dst[i] = (bf16_t)src[i];
}

// W [K][N] (row-major, x@W convention) -> Wt [N][K] bf16
__global__ __launch_bounds__(512) void cvt_transpose_w(const float* __restrict__ W,
                                                       bf16_t* __restrict__ Wt,
                                                       int K, int N) {
  int i = blockIdx.x * 512 + threadIdx.x;
  if (i < K * N) {
    int k = i / N, n = i % N;
    Wt[(size_t)n * K + k] = (bf16_t)W[i];
  }
}

// qkv [8192][3072] -> Q [bh][s][d] (pre-scaled by 1/sqrt(hd)) and K [bh][s][d].
__global__ __launch_bounds__(512) void rearrange_qk(const bf16_t* __restrict__ qkv,
                                                    bf16_t* __restrict__ Q,
                                                    bf16_t* __restrict__ Kd) {
  size_t i = (size_t)blockIdx.x * 512 + threadIdx.x;
  if (i >= (size_t)MROWS * 2 * DMODEL) return;
  int row = (int)(i / (2 * DMODEL));
  int j   = (int)(i % (2 * DMODEL));
  int b = row >> 11, s = row & (S_LEN - 1);
  int which = j >> 10, jj = j & (DMODEL - 1);
  int hh = jj >> 6, d = jj & 63;
  size_t bh = (size_t)b * NH + hh;
  bf16_t v = qkv[(size_t)row * (3 * DMODEL) + j];
  size_t dst = (bh * S_LEN + s) * HD + d;
  if (which == 0) Q[dst] = (bf16_t)(0.125f * (float)v);   // fold 1/sqrt(64), exact
  else            Kd[dst] = v;
}

// qkv V-slice -> Vt [bh][d][s]; indexed by OUTPUT position so stores coalesce.
__global__ __launch_bounds__(512) void rearrange_vt(const bf16_t* __restrict__ qkv,
                                                    bf16_t* __restrict__ Vt) {
  size_t i = (size_t)blockIdx.x * 512 + threadIdx.x;
  if (i >= (size_t)MROWS * DMODEL) return;
  int bh  = (int)(i >> 17);          // 64*2048 = 2^17 elements per (b,h)
  int rem = (int)(i & 131071);
  int d = rem >> 11, s = rem & (S_LEN - 1);
  int b = bh >> 4, hh = bh & 15;
  Vt[i] = qkv[((size_t)b * S_LEN + s) * (3 * DMODEL) + 2 * DMODEL + hh * 64 + d];
}

// ------------------------------- GEMM --------------------------------------
// C[M][N] = A[M][K] * Bt[N][K]^T + bias ; 8 waves/block, wave = 16x128 strip.

template <bool OUT_F32>
__global__ __launch_bounds__(256)
void gemm_bf16_wmma(const bf16_t* __restrict__ A, const bf16_t* __restrict__ Bt,
                    const float* __restrict__ bias, void* __restrict__ Cout,
                    int M, int N, int K) {
  const int lane = threadIdx.x & 31;
  const int wave = threadIdx.x >> 5;
  const int nlo  = lane & 15;
  const int h    = lane >> 4;
  const int m0 = blockIdx.y * 128 + wave * 16;
  const int n0 = blockIdx.x * 128;

  const bf16_t* arow = A + (size_t)(m0 + nlo) * K;
  const bf16_t* brow[8];
#pragma unroll
  for (int t = 0; t < 8; ++t) brow[t] = Bt + (size_t)(n0 + t * 16 + nlo) * K;

  v8f acc[8];
#pragma unroll
  for (int t = 0; t < 8; ++t) acc[t] = zero8();

#pragma unroll 2
  for (int k0 = 0; k0 < K; k0 += 32) {
    bf16v a = load_matA(arow, k0, h);      // reused by 8 WMMAs
#pragma unroll
    for (int t = 0; t < 8; ++t) {
      bf16v b = load_matB(brow[t], k0, h);
      acc[t] = wmma_bf16(a, b, acc[t]);
    }
  }

#pragma unroll
  for (int t = 0; t < 8; ++t) {
    const int n = n0 + t * 16 + nlo;
    const float bv = bias[n];
    if (OUT_F32) {
      float* C = (float*)Cout;
#pragma unroll
      for (int r = 0; r < 8; ++r)
        C[(size_t)(m0 + r + 8 * h) * N + n] = acc[t][r] + bv;
    } else {
      bf16_t* C = (bf16_t*)Cout;
#pragma unroll
      for (int r = 0; r < 8; ++r)
        C[(size_t)(m0 + r + 8 * h) * N + n] = (bf16_t)(acc[t][r] + bv);
    }
  }
}

// --------------------------- causal attention ------------------------------
// One wave = one 16-query tile of one (b,h). Online softmax over 64-key blocks
// (mask-free), 32-key masked tail for the diagonal. Row sums of P come from an
// extra WMMA against a ones matrix (D layout = per-row sums), so only the row
// max needs cross-lane shuffles.

__global__ __launch_bounds__(128)
void attn_fwd_wmma(const bf16_t* __restrict__ Q, const bf16_t* __restrict__ Km,
                   const bf16_t* __restrict__ Vt, bf16_t* __restrict__ O) {
  __shared__ __align__(16) bf16_t ldsP[4][16 * 64];   // per-wave P tile (D->A relayout)

  const int lane = threadIdx.x & 31;
  const int wave = threadIdx.x >> 5;
  const int nlo  = lane & 15;
  const int h    = lane >> 4;
  const int tile = blockIdx.x * 4 + wave;       // 8192 tiles total
  const int bh = tile >> 7;
  const int q0 = (tile & 127) * 16;
  const int bb = bh >> 4;
  const int hh = bh & 15;

  const bf16_t* Qb = Q  + (size_t)bh * S_LEN * HD;
  const bf16_t* Kb = Km + (size_t)bh * S_LEN * HD;
  const bf16_t* Vb = Vt + (size_t)bh * HD * S_LEN;

  const bf16_t* qrow = Qb + (size_t)(q0 + nlo) * HD;
  const bf16v aQ0 = load_matA(qrow, 0, h);      // dims 0..31 (Q pre-scaled)
  const bf16v aQ1 = load_matA(qrow, 32, h);     // dims 32..63
  const bf16v onesB = ones16();

  v8f o0 = zero8(), o1 = zero8(), o2 = zero8(), o3 = zero8();
  float mrow[8], lrow[8];
#pragma unroll
  for (int r = 0; r < 8; ++r) { mrow[r] = -1e30f; lrow[r] = 0.f; }

  bf16_t* myP = &ldsP[wave][0];

  // ---------------- mask-free 64-key main blocks ----------------
  int k0 = 0;
  for (; k0 + 64 <= q0; k0 += 64) {
    // prefetch next block's K rows / V rows (global_prefetch path)
    __builtin_prefetch(Kb + (size_t)(k0 + 64 + nlo) * HD, 0, 1);
    __builtin_prefetch(Vb + (size_t)nlo * S_LEN + k0 + 64, 0, 1);

    float p[4][8];
#pragma unroll
    for (int sub = 0; sub < 4; ++sub) {
      const bf16_t* krow = Kb + (size_t)(k0 + sub * 16 + nlo) * HD;
      v8f s = zero8();
      s = wmma_bf16(aQ0, load_matB(krow, 0, h), s);
      s = wmma_bf16(aQ1, load_matB(krow, 32, h), s);
#pragma unroll
      for (int r = 0; r < 8; ++r) p[sub][r] = s[r];
    }
    float fac[8];
#pragma unroll
    for (int r = 0; r < 8; ++r) {
      float e = fmaxf(fmaxf(p[0][r], p[1][r]), fmaxf(p[2][r], p[3][r]));
      float mn = fmaxf(mrow[r], rmax16(e));
      fac[r] = __expf(mrow[r] - mn);
      mrow[r] = mn;
    }
#pragma unroll
    for (int r = 0; r < 8; ++r) {
#pragma unroll
      for (int sub = 0; sub < 4; ++sub) {
        float pe = __expf(p[sub][r] - mrow[r]);
        myP[(r + 8 * h) * 64 + sub * 16 + nlo] = (bf16_t)pe;
      }
    }
    asm volatile("s_wait_dscnt 0" ::: "memory");
    const bf16v aP0 = load_matA(myP + nlo * 64, 0, h);   // keys k0+0..31
    const bf16v aP1 = load_matA(myP + nlo * 64, 32, h);  // keys k0+32..63
    // row sums of P via WMMA against ones (replaces shuffle-sum reduction)
    v8f rs = wmma_bf16(aP1, onesB, wmma_bf16(aP0, onesB, zero8()));
#pragma unroll
    for (int r = 0; r < 8; ++r) {
      lrow[r] = lrow[r] * fac[r] + rs[r];
      o0[r] *= fac[r]; o1[r] *= fac[r]; o2[r] *= fac[r]; o3[r] *= fac[r];
    }
#pragma unroll
    for (int d = 0; d < 4; ++d) {
      const bf16_t* vrow = Vb + (size_t)(d * 16 + nlo) * S_LEN + k0;
      v8f& o = (d == 0) ? o0 : (d == 1) ? o1 : (d == 2) ? o2 : o3;
      o = wmma_bf16(aP0, load_matB(vrow, 0, h), o);
      o = wmma_bf16(aP1, load_matB(vrow, 32, h), o);
    }
  }

  // ---------------- masked 32-key diagonal tail ----------------
  for (; k0 <= q0 + 15; k0 += 32) {
    float p0[8], p1[8];
    {
      const bf16_t* krow = Kb + (size_t)(k0 + nlo) * HD;
      v8f s = zero8();
      s = wmma_bf16(aQ0, load_matB(krow, 0, h), s);
      s = wmma_bf16(aQ1, load_matB(krow, 32, h), s);
#pragma unroll
      for (int r = 0; r < 8; ++r) p0[r] = s[r];
    }
    {
      const bf16_t* krow = Kb + (size_t)(k0 + 16 + nlo) * HD;
      v8f s = zero8();
      s = wmma_bf16(aQ0, load_matB(krow, 0, h), s);
      s = wmma_bf16(aQ1, load_matB(krow, 32, h), s);
#pragma unroll
      for (int r = 0; r < 8; ++r) p1[r] = s[r];
    }
    const int sk0 = k0 + nlo, sk1 = k0 + 16 + nlo;
    float fac[8];
#pragma unroll
    for (int r = 0; r < 8; ++r) {
      const int sq = q0 + r + 8 * h;
      if (sk0 > sq) p0[r] = -1e30f;
      if (sk1 > sq) p1[r] = -1e30f;
      float mn = fmaxf(mrow[r], rmax16(fmaxf(p0[r], p1[r])));
      fac[r] = __expf(mrow[r] - mn);
      mrow[r] = mn;
    }
#pragma unroll
    for (int r = 0; r < 8; ++r) {
      myP[(r + 8 * h) * 64 + nlo]      = (bf16_t)__expf(p0[r] - mrow[r]);
      myP[(r + 8 * h) * 64 + 16 + nlo] = (bf16_t)__expf(p1[r] - mrow[r]);
    }
    asm volatile("s_wait_dscnt 0" ::: "memory");
    const bf16v aP = load_matA(myP + nlo * 64, 0, h);
    v8f rs = wmma_bf16(aP, onesB, zero8());
#pragma unroll
    for (int r = 0; r < 8; ++r) {
      lrow[r] = lrow[r] * fac[r] + rs[r];
      o0[r] *= fac[r]; o1[r] *= fac[r]; o2[r] *= fac[r]; o3[r] *= fac[r];
    }
#pragma unroll
    for (int d = 0; d < 4; ++d) {
      const bf16_t* vrow = Vb + (size_t)(d * 16 + nlo) * S_LEN + k0;
      v8f& o = (d == 0) ? o0 : (d == 1) ? o1 : (d == 2) ? o2 : o3;
      o = wmma_bf16(aP, load_matB(vrow, 0, h), o);
    }
  }

  // normalize, write O[b][s][h*64+d] (row-major [8192][1024] for the out-proj GEMM)
  bf16_t* obase = O + (size_t)bb * S_LEN * DMODEL + hh * 64;
#pragma unroll
  for (int r = 0; r < 8; ++r) {
    const float inv = 1.0f / lrow[r];
    const size_t srow = (size_t)(q0 + r + 8 * h) * DMODEL;
    obase[srow + 0 * 16 + nlo] = (bf16_t)(o0[r] * inv);
    obase[srow + 1 * 16 + nlo] = (bf16_t)(o1[r] * inv);
    obase[srow + 2 * 16 + nlo] = (bf16_t)(o2[r] * inv);
    obase[srow + 3 * 16 + nlo] = (bf16_t)(o3[r] * inv);
  }
}

// ------------------------------- launch ------------------------------------

extern "C" void kernel_launch(void* const* d_in, const int* in_sizes, int n_in,
                              void* d_out, int out_size, void* d_ws, size_t ws_size,
                              hipStream_t stream) {
  (void)in_sizes; (void)n_in; (void)out_size; (void)ws_size;
  const float* x     = (const float*)d_in[0];
  const float* W_qkv = (const float*)d_in[1];
  const float* b_qkv = (const float*)d_in[2];
  const float* W_o   = (const float*)d_in[3];
  const float* b_o   = (const float*)d_in[4];
  float* out = (float*)d_out;

  const size_t PAD = 65536;   // tail over-read slack (values masked out)
  char* ws = (char*)d_ws;
  size_t off = 0;
  bf16_t* xb  = (bf16_t*)(ws + off); off += (size_t)MROWS * DMODEL * 2;
  bf16_t* Wtq = (bf16_t*)(ws + off); off += (size_t)3 * DMODEL * DMODEL * 2;
  bf16_t* Wto = (bf16_t*)(ws + off); off += (size_t)DMODEL * DMODEL * 2;
  bf16_t* qkv = (bf16_t*)(ws + off); off += (size_t)MROWS * 3 * DMODEL * 2;
  bf16_t* Qm  = (bf16_t*)(ws + off); off += (size_t)MROWS * DMODEL * 2;
  bf16_t* Kd  = (bf16_t*)(ws + off); off += (size_t)MROWS * DMODEL * 2 + PAD;
  bf16_t* Vt  = (bf16_t*)(ws + off); off += (size_t)MROWS * DMODEL * 2 + PAD;
  bf16_t* Ob  = (bf16_t*)(ws + off); off += (size_t)MROWS * DMODEL * 2 + PAD;

  // 1) precision conversion + weight transposes
  cvt_f32_bf16<<<(MROWS * DMODEL + 511) / 512, 512, 0, stream>>>(x, xb, MROWS * DMODEL);
  cvt_transpose_w<<<(3 * DMODEL * DMODEL + 511) / 512, 512, 0, stream>>>(W_qkv, Wtq, DMODEL, 3 * DMODEL);
  cvt_transpose_w<<<(DMODEL * DMODEL + 511) / 512, 512, 0, stream>>>(W_o, Wto, DMODEL, DMODEL);

  // 2) qkv = x @ W_qkv + b_qkv   (bf16 out)
  gemm_bf16_wmma<false><<<dim3(3 * DMODEL / 128, MROWS / 128), 256, 0, stream>>>(
      xb, Wtq, b_qkv, (void*)qkv, MROWS, 3 * DMODEL, DMODEL);

  // 3) head-major relayout (Q pre-scaled; K row-major; V transposed, coalesced out)
  rearrange_qk<<<((size_t)MROWS * 2 * DMODEL + 511) / 512, 512, 0, stream>>>(qkv, Qm, Kd);
  rearrange_vt<<<((size_t)MROWS * DMODEL + 511) / 512, 512, 0, stream>>>(qkv, Vt);

  // 4) causal flash attention (8192 query tiles / 4 waves per block)
  attn_fwd_wmma<<<BATCH * NH * (S_LEN / 16) / 4, 128, 0, stream>>>(Qm, Kd, Vt, Ob);

  // 5) out = O @ W_o + b_o   (fp32 out)
  gemm_bf16_wmma<true><<<dim3(DMODEL / 128, MROWS / 128), 256, 0, stream>>>(
      Ob, Wto, b_o, (void*)out, MROWS, DMODEL, DMODEL);
}
